// HybridQLSTM_65481071399679
// MI455X (gfx1250) — compile-verified, matrix operand order checked by reading, not run on hardware
//
#include <hip/hip_runtime.h>
#include <hip/hip_bf16.h>

// LSTM: T=512, B=128, D_in=512, D_h=512.
// Persistent kernel: 32 blocks x 8 waves run all 512 steps with a device-scope
// atomic barrier per step. Each wave owns one 16x16 (m,n) tile fused across
// all 4 gates. v_wmma_f32_16x16x32_bf16 over K=1024 with a 2-stage software
// pipeline. A-operand comes from a contiguous ping-pong comb[B][1024] buffer
// (x half | h half): single-base linear addressing, no per-iteration selects.
// x_{t+1} is staged into comb_next by the blocks during step t (copy hidden
// behind the WMMA loop); h_t is written into comb_next's h-half directly.

#define T_STEPS 512
#define BATCH   128
#define DIN     512
#define DH      512
#define KTOT    (DIN + DH)   // 1024
#define NGATES  4
#define NBLK    (DH / 16)    // 32 persistent blocks

typedef __attribute__((ext_vector_type(16))) __bf16 v16bf;
typedef __attribute__((ext_vector_type(8)))  float  v8f;

union BFrag {
    uint4 q[2];
    v16bf v;
};

struct Stage {
    float4 a0, a1, a2, a3;   // raw fp32 A tile (converted at consume time)
    BFrag  b[NGATES];        // bf16 B fragments, one per gate
};

__device__ __forceinline__ float sigmoid_f(float x) {
    return __builtin_amdgcn_rcpf(1.0f + __expf(-x));   // v_exp + v_rcp
}
__device__ __forceinline__ float tanh_f(float x) {
    float e = __expf(2.0f * x);
    return 1.0f - 2.0f * __builtin_amdgcn_rcpf(e + 1.0f);
}

// ---------------------------------------------------------------------------
// Pack weights: fp32 [K=1024, N=512] row-major -> bf16 [gate][N=512][K=1024]
// ---------------------------------------------------------------------------
__global__ __launch_bounds__(256) void wt_pack(
    const float* __restrict__ Wf, const float* __restrict__ Wi,
    const float* __restrict__ Wg, const float* __restrict__ Wo,
    unsigned short* __restrict__ wt)
{
    int idx = blockIdx.x * 256 + threadIdx.x;        // 0 .. 4*1024*512-1
    int g   = idx >> 19;
    int rem = idx & ((1 << 19) - 1);
    int k   = rem >> 9;
    int n   = rem & 511;
    const float* W = (g == 0) ? Wf : (g == 1) ? Wi : (g == 2) ? Wg : Wo;
    float v = W[(size_t)k * DH + n];
    __bf16 b = (__bf16)v;
    wt[(size_t)g * (KTOT * DH) + (size_t)n * KTOT + k] =
        __builtin_bit_cast(unsigned short, b);
}

// ---------------------------------------------------------------------------
// Init: comb0 = [x_0 | zeros], c = 0, barrier counters = 0.
// (ws is poisoned before timing; must re-init every call.)
// ---------------------------------------------------------------------------
__global__ __launch_bounds__(256) void init_state(
    const float* __restrict__ x,     // [T, B, DIN]
    float* __restrict__ comb0,       // [B, KTOT]
    float* __restrict__ c,           // [B, DH]
    unsigned int* __restrict__ ctr)  // [T]
{
    int i = blockIdx.x * 256 + threadIdx.x;          // 0 .. B*KTOT-1
    int row = i >> 10;                               // / KTOT
    int kc  = i & (KTOT - 1);
    comb0[i] = (kc < DIN) ? x[(size_t)row * DIN + kc] : 0.0f;
    if (i < BATCH * DH) c[i] = 0.0f;
    if (i < T_STEPS)    ctr[i] = 0u;
}

// ---------------------------------------------------------------------------
// Persistent LSTM.
// ---------------------------------------------------------------------------
__global__ __launch_bounds__(256)
__attribute__((amdgpu_waves_per_eu(1, 2)))
void lstm_persist(
    const float* __restrict__ x,        // [T, B, DIN]
    float*       __restrict__ stacked,  // [T, B, DH]
    float*       __restrict__ hx_out,   // [B, DH]
    float*       __restrict__ cx_out,   // [B, DH]
    const unsigned short* __restrict__ wt, // [4][DH][KTOT] bf16
    const float* __restrict__ bf_, const float* __restrict__ bi_,
    const float* __restrict__ bg_, const float* __restrict__ bo_,
    float* __restrict__ comb0, float* __restrict__ comb1, // ping-pong [B,KTOT]
    float* __restrict__ c_state,                          // [B, DH]
    unsigned int* __restrict__ ctr)                       // [T]
{
    const int lane    = threadIdx.x & 31;
    const int tile_m  = threadIdx.x >> 5;          // 0..7
    const int tile_n  = blockIdx.x;                // 0..31
    const int lhalf   = lane >> 4;                 // 0 or 1
    const int halfoff = lhalf * 8;
    const int col     = tile_n * 16 + (lane & 15);
    const int rowA    = tile_m * 16 + (lane & 15);

    const float bias_f = bf_[col];
    const float bias_i = bi_[col];
    const float bias_g = bg_[col];
    const float bias_o = bo_[col];

    const unsigned short* wrow[NGATES];
#pragma unroll
    for (int g = 0; g < NGATES; ++g)
        wrow[g] = wt + (size_t)g * (DH * KTOT) + (size_t)col * KTOT;

    // x-staging slice for this block: rows r0..r0+3, 2 float4 per thread.
    const int r0    = tile_n * 4;
    const int fidx0 = threadIdx.x;           // float4 index (0..255)
    const int fidx1 = threadIdx.x + 256;     // float4 index (256..511)

    for (int t = 0; t < T_STEPS; ++t) {
        const float* comb_cur  = (t & 1) ? comb1 : comb0;
        float*       comb_next = (t & 1) ? comb0 : comb1;
        float*       out_t     = stacked + (size_t)t * BATCH * DH;
        const float* xs        = x + (size_t)(t + 1) * BATCH * DIN;

        // Prefetch exactly the x_{t+1} lines this block will copy later.
        if (t + 1 < T_STEPS) {
            int row0 = fidx0 >> 7, colf0 = fidx0 & 127;
            int row1 = fidx1 >> 7, colf1 = fidx1 & 127;
            __builtin_prefetch(xs + (size_t)(r0 + row0) * DIN + colf0 * 4, 0, 3);
            __builtin_prefetch(xs + (size_t)(r0 + row1) * DIN + colf1 * 4, 0, 3);
        }

        v8f acc[NGATES];
#pragma unroll
        for (int e = 0; e < 8; ++e) {
            acc[0][e] = bias_f; acc[1][e] = bias_i;
            acc[2][e] = bias_g; acc[3][e] = bias_o;
        }

        // Single linear A base: comb row, like the weight rows.
        const float* ab = comb_cur + (size_t)rowA * KTOT + halfoff;

        auto stage_load = [&](Stage& s, int k) {
            const float* asrc = ab + k;
            s.a0 = *(const float4*)(asrc + 0);
            s.a1 = *(const float4*)(asrc + 4);
            s.a2 = *(const float4*)(asrc + 16);
            s.a3 = *(const float4*)(asrc + 20);
#pragma unroll
            for (int g = 0; g < NGATES; ++g) {
                const unsigned short* bp = wrow[g] + k + halfoff;
                s.b[g].q[0] = *(const uint4*)(bp);
                s.b[g].q[1] = *(const uint4*)(bp + 16);
            }
        };

        auto stage_mma = [&](Stage& s) {
            v16bf af;
            af[0]  = (__bf16)s.a0.x; af[1]  = (__bf16)s.a0.y;
            af[2]  = (__bf16)s.a0.z; af[3]  = (__bf16)s.a0.w;
            af[4]  = (__bf16)s.a1.x; af[5]  = (__bf16)s.a1.y;
            af[6]  = (__bf16)s.a1.z; af[7]  = (__bf16)s.a1.w;
            af[8]  = (__bf16)s.a2.x; af[9]  = (__bf16)s.a2.y;
            af[10] = (__bf16)s.a2.z; af[11] = (__bf16)s.a2.w;
            af[12] = (__bf16)s.a3.x; af[13] = (__bf16)s.a3.y;
            af[14] = (__bf16)s.a3.z; af[15] = (__bf16)s.a3.w;
#pragma unroll
            for (int g = 0; g < NGATES; ++g)
                acc[g] = __builtin_amdgcn_wmma_f32_16x16x32_bf16(
                    false, af, false, s.b[g].v, (short)0, acc[g], false, false);
        };

        // ---- 2-stage software pipeline over K = 1024 ----
        Stage s0, s1;
        stage_load(s0, 0);
#pragma unroll 1
        for (int kk = 0; kk < KTOT - 64; kk += 64) {
            stage_load(s1, kk + 32);
            stage_mma(s0);                 // covers s1's loads
            stage_load(s0, kk + 64);
            stage_mma(s1);                 // covers s0's loads
        }
        stage_load(s1, KTOT - 32);
        stage_mma(s0);
        stage_mma(s1);

        // ---- stage x_{t+1} into comb_next's x-half (prefetched above) ----
        if (t + 1 < T_STEPS) {
            int row0 = fidx0 >> 7, colf0 = fidx0 & 127;
            int row1 = fidx1 >> 7, colf1 = fidx1 & 127;
            float4 v0 = *(const float4*)(xs + (size_t)(r0 + row0) * DIN + colf0 * 4);
            float4 v1 = *(const float4*)(xs + (size_t)(r0 + row1) * DIN + colf1 * 4);
            *(float4*)(comb_next + (size_t)(r0 + row0) * KTOT + colf0 * 4) = v0;
            *(float4*)(comb_next + (size_t)(r0 + row1) * KTOT + colf1 * 4) = v1;
        }

        // ---- register-local cell update; h goes into comb_next h-half ----
#pragma unroll
        for (int e = 0; e < 8; ++e) {
            int row = tile_m * 16 + e + (lhalf << 3);
            size_t off = (size_t)row * DH + col;
            float fg = sigmoid_f(acc[0][e]);
            float ig = sigmoid_f(acc[1][e]);
            float gg = tanh_f   (acc[2][e]);
            float og = sigmoid_f(acc[3][e]);
            float cold = c_state[off];
            float cnew = fg * cold + ig * gg;
            float h    = og * tanh_f(cnew);
            c_state[off] = cnew;
            comb_next[(size_t)row * KTOT + DIN + col] = h;
            out_t[off] = h;                 // stacked[t] == h_t
            if (t == T_STEPS - 1) { hx_out[off] = h; cx_out[off] = cnew; }
        }

        // ---- device-scope barrier: all 32 blocks finish step t ----
        __threadfence();
        __syncthreads();
        if (threadIdx.x == 0) {
            __hip_atomic_fetch_add(&ctr[t], 1u, __ATOMIC_RELEASE,
                                   __HIP_MEMORY_SCOPE_AGENT);
            while (__hip_atomic_load(&ctr[t], __ATOMIC_ACQUIRE,
                                     __HIP_MEMORY_SCOPE_AGENT) < (unsigned)NBLK) {
                __builtin_amdgcn_s_sleep(1);
            }
        }
        __syncthreads();
        __builtin_amdgcn_fence(__ATOMIC_ACQUIRE, "agent");
    }
}

// ---------------------------------------------------------------------------
extern "C" void kernel_launch(void* const* d_in, const int* in_sizes, int n_in,
                              void* d_out, int out_size, void* d_ws, size_t ws_size,
                              hipStream_t stream)
{
    (void)in_sizes; (void)n_in; (void)out_size; (void)ws_size;

    const float* x   = (const float*)d_in[0];
    const float* Wf  = (const float*)d_in[1];
    const float* bf_ = (const float*)d_in[2];
    const float* Wi  = (const float*)d_in[3];
    const float* bi_ = (const float*)d_in[4];
    const float* Wg  = (const float*)d_in[5];
    const float* bg_ = (const float*)d_in[6];
    const float* Wo  = (const float*)d_in[7];
    const float* bo_ = (const float*)d_in[8];

    float* out     = (float*)d_out;
    float* stacked = out;
    float* hx      = out + (size_t)T_STEPS * BATCH * DH;
    float* cx      = hx  + (size_t)BATCH * DH;

    // Workspace: 4MB weights + 2x512KB comb + 256KB c + 2KB counters
    char* ws = (char*)d_ws;
    unsigned short* wt = (unsigned short*)ws;
    size_t wt_bytes = (size_t)NGATES * DH * KTOT * sizeof(unsigned short);
    float* comb0 = (float*)(ws + wt_bytes);          // [B, KTOT]
    float* comb1 = comb0 + (size_t)BATCH * KTOT;     // [B, KTOT]
    float* c     = comb1 + (size_t)BATCH * KTOT;     // [B, DH]
    unsigned int* ctr = (unsigned int*)(c + (size_t)BATCH * DH);

    wt_pack<<<(NGATES * KTOT * DH) / 256, 256, 0, stream>>>(Wf, Wi, Wg, Wo, wt);
    init_state<<<(BATCH * KTOT) / 256, 256, 0, stream>>>(x, comb0, c, ctr);
    lstm_persist<<<NBLK, 256, 0, stream>>>(
        x, stacked, hx, cx, wt, bf_, bi_, bg_, bo_, comb0, comb1, c, ctr);
}
